// AttentionWithCoverage_17626545783256
// MI455X (gfx1250) — matrix operand order repaired and checked by reading, not compiled, not run on hardware
//
#include <hip/hip_runtime.h>
#include <hip/hip_bf16.h>
#include <math.h>

typedef __attribute__((ext_vector_type(16))) __bf16        v16bf;
typedef __attribute__((ext_vector_type(8)))  float         v8f;
typedef __attribute__((ext_vector_type(4)))  float         v4f;
typedef __attribute__((ext_vector_type(8)))  unsigned int  v8u;

#define BSZ 32
#define LSEQ 2048
#define DIM 1024
#define NEG_INF_F (-1e30f)

// pack two f32 into one dword of two bf16 (round-to-nearest-even), a -> low 16
__device__ __forceinline__ unsigned int packbf(float a, float b) {
  unsigned ua = __builtin_bit_cast(unsigned, a);
  unsigned ub = __builtin_bit_cast(unsigned, b);
  ua = ua + 0x7FFFu + ((ua >> 16) & 1u);
  ub = ub + 0x7FFFu + ((ub >> 16) & 1u);
  return (ua >> 16) | (ub & 0xFFFF0000u);
}

// ---------------------------------------------------------------------------
// K0: pack U [1024,1024] f32 (row-major [k][n]) into WMMA B-fragment order.
// Fragment for (nt, kk): lane = column-in-tile (lane&15), lane<16 -> K half 0,
// lane>=16 -> K half 1; 16 consecutive K values packed 2-per-dword per lane.
// Storage: Upk[((nt*32 + kk)*32 + lane)*8 + j]  (dwords, 32B per lane-chunk)
// ---------------------------------------------------------------------------
__global__ void pack_u_kernel(const float* __restrict__ U, unsigned int* __restrict__ Upk) {
  int t = blockIdx.x * 256 + threadIdx.x;        // 0 .. 65535
  int lane = t & 31;
  int kk   = (t >> 5) & 31;
  int nt   = t >> 10;
  int n  = nt * 16 + (lane & 15);
  int k0 = kk * 32 + ((lane < 16) ? 0 : 16);
  unsigned int* dst = Upk + (size_t)t * 8;
#pragma unroll
  for (int j = 0; j < 8; ++j) {
    float f0 = U[(size_t)(k0 + 2 * j) * DIM + n];
    float f1 = U[(size_t)(k0 + 2 * j + 1) * DIM + n];
    dst[j] = packbf(f0, f1);
  }
}

// ---------------------------------------------------------------------------
// K1: vW[b, j] = sum_d vector[b, d] * W[d, j]
// ---------------------------------------------------------------------------
__global__ void vw_kernel(const float* __restrict__ vec, const float* __restrict__ W,
                          float* __restrict__ vWg) {
  int b = blockIdx.y;
  int j = blockIdx.x * 256 + threadIdx.x;
  __shared__ float vsh[DIM];
  for (int i = threadIdx.x; i < DIM; i += 256) vsh[i] = vec[b * DIM + i];
  __syncthreads();
  float acc = 0.f;
  for (int d = 0; d < DIM; ++d) acc = fmaf(vsh[d], W[(size_t)d * DIM + j], acc);
  vWg[b * DIM + j] = acc;
}

// ---------------------------------------------------------------------------
// K2: fused  sims_raw[b,l] = dot(tanh(vW[b]+matrix[b,l,:]@U+cov*w1), v)
// One workgroup (8 waves) per (batch, 32-row L-tile): two 16-row A tiles per
// K-step share one B fragment -> halves L2 traffic on packed U.
// bf16 WMMA, f32 accumulate.
// ---------------------------------------------------------------------------
__global__ void __launch_bounds__(256)
sims_kernel(const float* __restrict__ matrix, const float* __restrict__ coverage,
            const unsigned int* __restrict__ Upk, const float* __restrict__ vWg,
            const float* __restrict__ vvec, const float* __restrict__ w1,
            float* __restrict__ sims_raw) {
  const int lt = blockIdx.x;        // 0..63 (32-row tiles)
  const int b  = blockIdx.y;        // 0..31
  const int tid = threadIdx.x;
  const int l0 = lt * 32;

  // A fragments for both 16-row tiles, all 32 K-steps: 64 KB
  __shared__ __align__(32) unsigned int lds_a[2 * 32 * 32 * 8];
  __shared__ float vwsh[DIM];
  __shared__ float w1sh[DIM];
  __shared__ float vsh[DIM];
  __shared__ float covsh[32];
  __shared__ float simspart[8][32];

  // ---- stage matrix tile as packed bf16 A-fragments (contiguous k = 16*half + s, same order as packed U) ----
  const float* mbase = matrix + ((size_t)b * LSEQ + l0) * DIM;
#pragma unroll
  for (int i = 0; i < 8; ++i) {
    int c    = tid + i * 256;       // 0..2047 = tile*1024 + kk*32 + lane
    int tile = c >> 10;
    int cc   = c & 1023;
    int kk = cc >> 5;
    int ln = cc & 31;
    int row = tile * 16 + (ln & 15);
    int kb = kk * 32 + ((ln < 16) ? 0 : 16);   // first 8-K chunk
    int k2 = kk * 32 + ((ln < 16) ? 8 : 24);   // second 8-K chunk
    const float* mr = mbase + (size_t)row * DIM;
    v4f f0 = __builtin_nontemporal_load((const v4f*)(mr + kb));
    v4f f1 = __builtin_nontemporal_load((const v4f*)(mr + kb + 4));
    v4f f2 = __builtin_nontemporal_load((const v4f*)(mr + k2));
    v4f f3 = __builtin_nontemporal_load((const v4f*)(mr + k2 + 4));
    unsigned int* dst = &lds_a[c * 8];
    dst[0] = packbf(f0.x, f0.y); dst[1] = packbf(f0.z, f0.w);
    dst[2] = packbf(f1.x, f1.y); dst[3] = packbf(f1.z, f1.w);
    dst[4] = packbf(f2.x, f2.y); dst[5] = packbf(f2.z, f2.w);
    dst[6] = packbf(f3.x, f3.y); dst[7] = packbf(f3.z, f3.w);
  }
  *(v4f*)&vwsh[tid * 4] = *(const v4f*)&vWg[b * DIM + tid * 4];
  *(v4f*)&w1sh[tid * 4] = *(const v4f*)&w1[tid * 4];
  *(v4f*)&vsh[tid * 4]  = *(const v4f*)&vvec[tid * 4];
  if (tid < 32) covsh[tid] = coverage[b * LSEQ + l0 + tid];
  __syncthreads();

  const int wid  = tid >> 5;
  const int lane = tid & 31;
  const int ln16 = lane & 15;
  const int rowbase = (lane < 16) ? 0 : 8;

  float srow[16];
#pragma unroll
  for (int r = 0; r < 16; ++r) srow[r] = 0.f;

  const unsigned int* ap0 = &lds_a[lane * 8];            // tile 0
  const unsigned int* ap1 = &lds_a[(1024 + lane) * 8];   // tile 1
  for (int nti = 0; nti < 8; ++nti) {
    int nt = wid * 8 + nti;                              // this wave's N-tile
    v8f acc0 = {0.f, 0.f, 0.f, 0.f, 0.f, 0.f, 0.f, 0.f};
    v8f acc1 = {0.f, 0.f, 0.f, 0.f, 0.f, 0.f, 0.f, 0.f};
    const v8u* bp = ((const v8u*)Upk) + ((size_t)nt * 32 * 32 + lane);
#pragma unroll 4
    for (int kk = 0; kk < 32; ++kk) {
      v8u bu  = bp[(size_t)kk * 32];                     // global: 2x b128, L2-hot
      v8u au0 = *(const v8u*)(ap0 + (size_t)kk * 256);   // LDS: 2x ds_load_b128
      v8u au1 = *(const v8u*)(ap1 + (size_t)kk * 256);
      v16bf bb = __builtin_bit_cast(v16bf, bu);
      v16bf a0 = __builtin_bit_cast(v16bf, au0);
      v16bf a1 = __builtin_bit_cast(v16bf, au1);
      acc0 = __builtin_amdgcn_wmma_f32_16x16x32_bf16(
          false, a0, false, bb, (short)0, acc0, false, false);
      acc1 = __builtin_amdgcn_wmma_f32_16x16x32_bf16(
          false, a1, false, bb, (short)0, acc1, false, false);
    }
    // epilogue: C layout -> VGPR r holds row (rowbase+r), column nt*16 + ln16
    int col = nt * 16 + ln16;
    float vw  = vwsh[col];
    float w1v = w1sh[col];
    float vv  = vsh[col];
#pragma unroll
    for (int r = 0; r < 8; ++r) {
      float v0 = acc0[r] + vw + covsh[rowbase + r] * w1v;        // rows l0+0..15
      float v1 = acc1[r] + vw + covsh[16 + rowbase + r] * w1v;   // rows l0+16..31
      srow[r]     += tanhf(v0) * vv;
      srow[8 + r] += tanhf(v1) * vv;
    }
  }

  // reduce across 16 columns (xor masks stay inside each lane half)
#pragma unroll
  for (int r = 0; r < 16; ++r) {
    float s = srow[r];
    s += __shfl_xor(s, 1, 32);
    s += __shfl_xor(s, 2, 32);
    s += __shfl_xor(s, 4, 32);
    s += __shfl_xor(s, 8, 32);
    srow[r] = s;
  }
  if (lane == 0) {
#pragma unroll
    for (int r = 0; r < 8; ++r) {
      simspart[wid][r]      = srow[r];        // rows 0..7
      simspart[wid][16 + r] = srow[8 + r];    // rows 16..23
    }
  } else if (lane == 16) {
#pragma unroll
    for (int r = 0; r < 8; ++r) {
      simspart[wid][8 + r]  = srow[r];        // rows 8..15
      simspart[wid][24 + r] = srow[8 + r];    // rows 24..31
    }
  }
  __syncthreads();
  if (tid < 32) {
    float s = 0.f;
#pragma unroll
    for (int w = 0; w < 8; ++w) s += simspart[w][tid];
    sims_raw[b * LSEQ + l0 + tid] = s;
  }
}

// ---------------------------------------------------------------------------
// K3: masked softmax over L per batch + coverage_out
// ---------------------------------------------------------------------------
__global__ void softmax_kernel(const float* __restrict__ sims_raw,
                               const unsigned char* __restrict__ mask,
                               const float* __restrict__ coverage,
                               float* __restrict__ sims_out,
                               float* __restrict__ cov_out) {
  int b = blockIdx.x;
  int tid = threadIdx.x;
  __shared__ float red[256];
  float x[8];
  float lm = -INFINITY;
#pragma unroll
  for (int i = 0; i < 8; ++i) {
    int l = tid + i * 256;
    bool mk = mask[b * LSEQ + l] != 0;
    x[i] = mk ? sims_raw[b * LSEQ + l] : NEG_INF_F;
    lm = fmaxf(lm, x[i]);
  }
  red[tid] = lm;
  __syncthreads();
  for (int s = 128; s > 0; s >>= 1) {
    if (tid < s) red[tid] = fmaxf(red[tid], red[tid + s]);
    __syncthreads();
  }
  float m = red[0];
  __syncthreads();
  float e[8];
  float ls = 0.f;
#pragma unroll
  for (int i = 0; i < 8; ++i) { e[i] = __expf(x[i] - m); ls += e[i]; }
  red[tid] = ls;
  __syncthreads();
  for (int s = 128; s > 0; s >>= 1) {
    if (tid < s) red[tid] += red[tid + s];
    __syncthreads();
  }
  float inv = 1.f / red[0];
#pragma unroll
  for (int i = 0; i < 8; ++i) {
    int l = tid + i * 256;
    bool mk = mask[b * LSEQ + l] != 0;
    float p = mk ? e[i] * inv : 0.f;
    sims_out[b * LSEQ + l] = p;
    cov_out[b * LSEQ + l] = coverage[b * LSEQ + l] + p;
  }
}

// ---------------------------------------------------------------------------
// K4: weighted[b,d] = sum_l sims[b,l] * matrix[b,l,d]   (bandwidth bound)
// ---------------------------------------------------------------------------
__global__ void weighted_kernel(const float* __restrict__ sims,
                                const float* __restrict__ matrix,
                                float* __restrict__ out) {
  int b  = blockIdx.y;
  int cz = blockIdx.x;
  int tid = threadIdx.x;
  __shared__ float ssim[LSEQ];
  for (int i = tid; i < LSEQ; i += 256) ssim[i] = sims[b * LSEQ + i];
  __syncthreads();
  int c = cz * 256 + tid;
  const float* mb = matrix + (size_t)b * LSEQ * DIM + c;
  float acc = 0.f;
#pragma unroll 4
  for (int l = 0; l < LSEQ; ++l) {
    float m = __builtin_nontemporal_load(mb + (size_t)l * DIM);
    acc = fmaf(ssim[l], m, acc);
  }
  out[b * DIM + c] = acc;
}

// ---------------------------------------------------------------------------
extern "C" void kernel_launch(void* const* d_in, const int* in_sizes, int n_in,
                              void* d_out, int out_size, void* d_ws, size_t ws_size,
                              hipStream_t stream) {
  const float*         vector   = (const float*)d_in[0];
  const float*         matrix   = (const float*)d_in[1];
  const unsigned char* mmask    = (const unsigned char*)d_in[2];
  const float*         coverage = (const float*)d_in[3];
  const float*         W        = (const float*)d_in[4];
  const float*         U        = (const float*)d_in[5];
  const float*         vvec     = (const float*)d_in[6];
  const float*         w1       = (const float*)d_in[7];

  float* out      = (float*)d_out;
  float* weighted = out;                      // [32,1024]
  float* sims_out = out + BSZ * DIM;          // [32,2048]
  float* cov_out  = sims_out + BSZ * LSEQ;    // [32,2048]

  // workspace layout
  unsigned int* Upk = (unsigned int*)d_ws;                                   // 2 MB bf16-packed U
  float* vWg        = (float*)((char*)d_ws + (size_t)2 * 1024 * 1024);       // 128 KB
  float* sims_raw   = (float*)((char*)d_ws + (size_t)2 * 1024 * 1024 + 128 * 1024); // 256 KB

  pack_u_kernel<<<256, 256, 0, stream>>>(U, Upk);
  vw_kernel<<<dim3(4, BSZ), 256, 0, stream>>>(vector, W, vWg);
  sims_kernel<<<dim3(LSEQ / 32, BSZ), 256, 0, stream>>>(matrix, coverage, Upk, vWg,
                                                        vvec, w1, sims_raw);
  softmax_kernel<<<BSZ, 256, 0, stream>>>(sims_raw, mmask, coverage, sims_out, cov_out);
  weighted_kernel<<<dim3(4, BSZ), 256, 0, stream>>>(sims_out, matrix, weighted);
}
